// Mamba_40243843563850
// MI455X (gfx1250) — compile-verified, matrix operand order checked by reading, not run on hardware
//
#include <hip/hip_runtime.h>
#include <hip/hip_bf16.h>

// ---------------------------------------------------------------------------
// Mamba forward for MI455X (gfx1250, wave32, WMMA + async global->LDS).
//   hs(2,2048,1024) -> xz GEMM -> conv+SiLU -> x_dbl GEMM -> delta GEMM+softplus
//   -> selective scan (fused SiLU(z), D-skip) -> out GEMM.
// GEMMs: v_wmma_f32_16x16x32_f16, f32 accumulate.  A and B tiles both staged
// in double-buffered LDS via GLOBAL_LOAD_ASYNC_TO_LDS_B128 (ASYNCcnt), so the
// steady-state inner loop touches only LDS while async copies prefetch the
// next K-chunk.
// ---------------------------------------------------------------------------

typedef _Float16 f16;
typedef __attribute__((ext_vector_type(16))) _Float16 v16h;
typedef __attribute__((ext_vector_type(8)))  _Float16 v8h;
typedef __attribute__((ext_vector_type(8)))  float    v8f;
typedef __attribute__((ext_vector_type(4)))  int      v4i;

#define D_MODEL 1024
#define D_INNER 2048
#define D_STATE 16
#define DT_RANK 64
#define NXD     96      // DT_RANK + 2*D_STATE
#define BATCH   2
#define SEQLEN  2048
#define MROWS   (BATCH * SEQLEN)   // 4096

#define KC   64          // K-chunk staged in LDS per stage
#define LDSP 72          // padded row pitch (halfs) to spread LDS banks

#define AS_GLOBAL __attribute__((address_space(1)))
#define AS_SHARED __attribute__((address_space(3)))

// ---- CDNA5 async global->LDS copy (ASYNCcnt) -------------------------------

#if __has_builtin(__builtin_amdgcn_global_load_async_to_lds_b128)
#define HAVE_ASYNC_LDS 1
#else
#define HAVE_ASYNC_LDS 0
#endif

static __device__ __forceinline__ void async_copy_b128(const f16* g, f16* l) {
#if HAVE_ASYNC_LDS
  __builtin_amdgcn_global_load_async_to_lds_b128(
      (AS_GLOBAL v4i*)g, (AS_SHARED v4i*)l, 0, 0);
#else
  *(v8h*)l = *(const v8h*)g;     // synchronous fallback (ds_store_b128)
#endif
}

static __device__ __forceinline__ void async_wait0() {
#if HAVE_ASYNC_LDS
  asm volatile("s_wait_asynccnt 0x0" ::: "memory");
#endif
}

// ---- WMMA helpers ----------------------------------------------------------

static __device__ __forceinline__ v8f wmma_f16(v16h a, v16h b, v8f c) {
  // 8 args: (neg_a, A, neg_b, B, c_mod, C, reuse_a, reuse_b)
  return __builtin_amdgcn_wmma_f32_16x16x32_f16(false, a, false, b, (short)0, c,
                                                false, false);
}

// Per-lane fragment load for 16-bit 16x32 A (or 32x16 B from a transposed NxK
// matrix).  p points at element [row][k0 + 8*h]; second half-group at +16.
static __device__ __forceinline__ v16h load_frag16(const f16* p) {
  v8h lo = *reinterpret_cast<const v8h*>(p);
  v8h hi = *reinterpret_cast<const v8h*>(p + 16);
  v16h r;
#pragma unroll
  for (int i = 0; i < 8; ++i) { r[i] = lo[i]; r[8 + i] = hi[i]; }
  return r;
}

static __device__ __forceinline__ v8f vzero8() {
  v8f z = {0.f, 0.f, 0.f, 0.f, 0.f, 0.f, 0.f, 0.f};
  return z;
}

static __device__ __forceinline__ float silu_f(float v) {
  return v / (1.f + __expf(-v));
}

// ---- pack / convert kernels ------------------------------------------------

__global__ __launch_bounds__(256) void k_pack_hs(const float* __restrict__ src,
                                                 f16* __restrict__ dst, int n) {
  int i = blockIdx.x * 256 + threadIdx.x;
  if (i < n) dst[i] = (f16)src[i];
}

// W_in (1024 x 4096) -> WxT (2048 x 1024), WzT (2048 x 1024): deinterleave+T
__global__ __launch_bounds__(256) void k_pack_win(const float* __restrict__ w,
                                                  f16* __restrict__ wxT,
                                                  f16* __restrict__ wzT) {
  int i = blockIdx.x * 256 + threadIdx.x;          // i = n*1024 + k
  int nn = i >> 10, kk = i & 1023;
  wxT[i] = (f16)w[kk * (2 * D_INNER) + 2 * nn];
  wzT[i] = (f16)w[kk * (2 * D_INNER) + 2 * nn + 1];
}

// W_x (2048 x 96) -> WxlT (96 x 2048)
__global__ __launch_bounds__(256) void k_pack_wx(const float* __restrict__ w,
                                                 f16* __restrict__ wT) {
  int i = blockIdx.x * 256 + threadIdx.x;          // i = n*2048 + k
  int nn = i >> 11, kk = i & 2047;
  wT[i] = (f16)w[kk * NXD + nn];
}

// W_dt (64 x 2048) -> WdtT (2048 x 64)
__global__ __launch_bounds__(256) void k_pack_wdt(const float* __restrict__ w,
                                                  f16* __restrict__ wT) {
  int i = blockIdx.x * 256 + threadIdx.x;          // i = n*64 + k
  int nn = i >> 6, kk = i & 63;
  wT[i] = (f16)w[kk * D_INNER + nn];
}

// W_out (2048 x 1024) -> WoutT (1024 x 2048)
__global__ __launch_bounds__(256) void k_pack_wout(const float* __restrict__ w,
                                                   f16* __restrict__ wT) {
  int i = blockIdx.x * 256 + threadIdx.x;          // i = n*2048 + k
  int nn = i >> 11, kk = i & 2047;
  wT[i] = (f16)w[kk * D_MODEL + nn];
}

// ---- generic GEMM: C(f32, MxN) = A(f16, MxK) * BT(f16, NxK)^T --------------
// Block: 256 threads = 8 waves as 4(M) x 2(N); block tile 64M x 128N; each
// wave computes 16M x 64N (4 accumulators).  A tile (64xKC) and B tile
// (128xKC) staged in double-buffered LDS via async copies; inner loop reads
// only LDS.
__global__ __launch_bounds__(256) void k_gemm_nt(const f16* __restrict__ A,
                                                 const f16* __restrict__ BT,
                                                 float* __restrict__ C,
                                                 int K, int N) {
  __shared__ f16 sA[2][64][LDSP];
  __shared__ f16 sB[2][128][LDSP];

  int tid = threadIdx.x, wave = tid >> 5, lane = tid & 31;
  int mw = wave & 3, nw = wave >> 2;
  int m0 = blockIdx.x * 64;
  int n0 = blockIdx.y * 128;
  int mr = lane & 15, h = lane >> 4;

  v8f acc[4];
#pragma unroll
  for (int t = 0; t < 4; ++t) acc[t] = vzero8();

  // A: 64 rows x 8 b128-chunks = 512 -> 2 per thread.
  // B: 128 rows x 8 b128-chunks = 1024 -> 4 per thread.
  auto stage = [&](int buf, int kc) {
#pragma unroll
    for (int j = 0; j < 2; ++j) {
      int c = tid + 256 * j;
      int row = c >> 3, col = (c & 7) * 8;
      async_copy_b128(A + (size_t)(m0 + row) * K + kc + col, &sA[buf][row][col]);
    }
#pragma unroll
    for (int j = 0; j < 4; ++j) {
      int c = tid + 256 * j;
      int row = c >> 3, col = (c & 7) * 8;
      async_copy_b128(BT + (size_t)(n0 + row) * K + kc + col, &sB[buf][row][col]);
    }
  };

  stage(0, 0);
  int nstage = K / KC;
  for (int s = 0; s < nstage; ++s) {
    async_wait0();
    __syncthreads();
    if (s + 1 < nstage) stage((s + 1) & 1, (s + 1) * KC);

#pragma unroll
    for (int kk = 0; kk < KC; kk += 32) {
      v16h a = load_frag16(&sA[s & 1][mw * 16 + mr][kk + 8 * h]);
      v16h b[4];
#pragma unroll
      for (int t = 0; t < 4; ++t)
        b[t] = load_frag16(&sB[s & 1][nw * 64 + t * 16 + mr][kk + 8 * h]);
#pragma unroll
      for (int t = 0; t < 4; ++t) acc[t] = wmma_f16(a, b[t], acc[t]);
    }
  }
#pragma unroll
  for (int t = 0; t < 4; ++t)
#pragma unroll
    for (int r = 0; r < 8; ++r) {
      int row = m0 + mw * 16 + r + 8 * h;
      int col = n0 + nw * 64 + t * 16 + (lane & 15);
      C[(size_t)row * N + col] = acc[t][r];
    }
}

// ---- depthwise conv(4, SAME: pad_lo=1) + bias + SiLU -----------------------
__global__ __launch_bounds__(256) void k_conv(const float* __restrict__ xpre,
                                              const float* __restrict__ ck,
                                              const float* __restrict__ cb,
                                              float* __restrict__ xf,
                                              f16* __restrict__ x16) {
  int i = blockIdx.x * 256 + threadIdx.x;          // i = m*2048 + d
  int m = i >> 11, d = i & 2047;
  int b = m >> 11, l = m & 2047;
  float s = cb[d];
#pragma unroll
  for (int w = 0; w < 4; ++w) {
    int ll = l + w - 1;
    if (ll >= 0 && ll < SEQLEN)
      s += ck[w * D_INNER + d] * xpre[(b * SEQLEN + ll) * D_INNER + d];
  }
  s = silu_f(s);
  xf[i]  = s;
  x16[i] = (f16)s;
}

// ---- GEMM 2: x_dbl = x @ W_x; split into dt16 / Bm / Cm --------------------
// M=4096, K=2048, N=96.  8 waves stacked in M (128M); A tile (128xKC) and B
// tile (96xKC) both staged in double-buffered LDS.
__global__ __launch_bounds__(256) void k_gemm_xdbl(const f16* __restrict__ A,
                                                   const f16* __restrict__ BT,
                                                   f16* __restrict__ dt16,
                                                   float* __restrict__ Bm,
                                                   float* __restrict__ Cm) {
  __shared__ f16 sA[2][128][LDSP];
  __shared__ f16 sB[2][96][LDSP];

  int tid = threadIdx.x, wave = tid >> 5, lane = tid & 31;
  int m0 = blockIdx.x * 128;
  int mr = lane & 15, h = lane >> 4;

  v8f acc[6];
#pragma unroll
  for (int t = 0; t < 6; ++t) acc[t] = vzero8();

  // A: 128 rows x 8 = 1024 chunks -> 4/thread; B: 96 x 8 = 768 -> 3/thread.
  auto stage = [&](int buf, int kc) {
#pragma unroll
    for (int j = 0; j < 4; ++j) {
      int c = tid + 256 * j;
      int row = c >> 3, col = (c & 7) * 8;
      async_copy_b128(A + (size_t)(m0 + row) * D_INNER + kc + col,
                      &sA[buf][row][col]);
    }
#pragma unroll
    for (int j = 0; j < 3; ++j) {
      int c = tid + 256 * j;
      int row = c >> 3, col = (c & 7) * 8;
      async_copy_b128(BT + (size_t)row * D_INNER + kc + col, &sB[buf][row][col]);
    }
  };

  stage(0, 0);
  for (int s = 0; s < D_INNER / KC; ++s) {
    async_wait0();
    __syncthreads();
    if (s + 1 < D_INNER / KC) stage((s + 1) & 1, (s + 1) * KC);

#pragma unroll
    for (int kk = 0; kk < KC; kk += 32) {
      v16h a = load_frag16(&sA[s & 1][wave * 16 + mr][kk + 8 * h]);
      v16h b[6];
#pragma unroll
      for (int t = 0; t < 6; ++t)
        b[t] = load_frag16(&sB[s & 1][t * 16 + mr][kk + 8 * h]);
#pragma unroll
      for (int t = 0; t < 6; ++t) acc[t] = wmma_f16(a, b[t], acc[t]);
    }
  }
#pragma unroll
  for (int t = 0; t < 6; ++t)
#pragma unroll
    for (int r = 0; r < 8; ++r) {
      int row = m0 + wave * 16 + r + 8 * h;
      int cl  = lane & 15;
      float v = acc[t][r];
      if (t < 4)        dt16[row * DT_RANK + t * 16 + cl] = (f16)v;
      else if (t == 4)  Bm[row * D_STATE + cl] = v;
      else              Cm[row * D_STATE + cl] = v;
    }
}

// ---- GEMM 3: delta = softplus(dt @ W_dt + b_dt) ----------------------------
// M=4096, K=64, N=2048.  K tiny: direct loads (B panel is 256KB, L2 resident).
__global__ __launch_bounds__(256) void k_gemm_delta(const f16* __restrict__ A,
                                                    const f16* __restrict__ BT,
                                                    const float* __restrict__ bdt,
                                                    float* __restrict__ delta) {
  int tid = threadIdx.x, wave = tid >> 5, lane = tid & 31;
  int m0 = blockIdx.x * 128 + wave * 16;
  int n0 = blockIdx.y * 64;
  int mr = lane & 15, h = lane >> 4;

  v8f acc[4];
#pragma unroll
  for (int t = 0; t < 4; ++t) acc[t] = vzero8();

#pragma unroll
  for (int k0 = 0; k0 < DT_RANK; k0 += 32) {
    v16h a = load_frag16(A + (m0 + mr) * DT_RANK + k0 + 8 * h);
    v16h b[4];
#pragma unroll
    for (int t = 0; t < 4; ++t)
      b[t] = load_frag16(BT + (n0 + t * 16 + mr) * DT_RANK + k0 + 8 * h);
#pragma unroll
    for (int t = 0; t < 4; ++t) acc[t] = wmma_f16(a, b[t], acc[t]);
  }
#pragma unroll
  for (int t = 0; t < 4; ++t)
#pragma unroll
    for (int r = 0; r < 8; ++r) {
      int row = m0 + r + 8 * h;
      int col = n0 + t * 16 + (lane & 15);
      float v = acc[t][r] + bdt[col];
      float sp = (v > 20.f) ? v : log1pf(__expf(v));
      delta[row * D_INNER + col] = sp;
    }
}

// ---- selective scan + fused D-skip, SiLU(z), f16 convert -------------------
__global__ __launch_bounds__(256) void k_scan(const float* __restrict__ delta,
                                              const float* __restrict__ xf,
                                              const float* __restrict__ Bm,
                                              const float* __restrict__ Cm,
                                              const float* __restrict__ zf,
                                              const float* __restrict__ A_log,
                                              const float* __restrict__ Dp,
                                              f16* __restrict__ ypre16) {
  int tid = threadIdx.x;
  int b   = blockIdx.x >> 3;
  int d   = (blockIdx.x & 7) * 256 + tid;

  float st[D_STATE], Arow[D_STATE];
#pragma unroll
  for (int n = 0; n < D_STATE; ++n) {
    st[n]   = 0.f;
    Arow[n] = -__expf(A_log[d * D_STATE + n]);
  }
  float Dd = Dp[d];

  __shared__ float sB[16][D_STATE];
  __shared__ float sC[16][D_STATE];

  for (int l0 = 0; l0 < SEQLEN; l0 += 16) {
    {
      int t = tid >> 4, n = tid & 15;                 // 256 == 16*16
      int row = b * SEQLEN + l0 + t;
      sB[t][n] = Bm[row * D_STATE + n];
      sC[t][n] = Cm[row * D_STATE + n];
    }
    __syncthreads();
#pragma unroll 4
    for (int t = 0; t < 16; ++t) {
      int row  = b * SEQLEN + l0 + t;
      float dv = delta[row * D_INNER + d];
      float xv = xf[row * D_INNER + d];
      float dx = dv * xv;
      float y  = 0.f;
#pragma unroll
      for (int n = 0; n < D_STATE; ++n) {
        float dA = __expf(dv * Arow[n]);
        st[n] = dA * st[n] + dx * sB[t][n];
        y += st[n] * sC[t][n];
      }
      float zv = zf[row * D_INNER + d];
      float o  = (y + xv * Dd) * silu_f(zv);
      ypre16[row * D_INNER + d] = (f16)o;
    }
    __syncthreads();
  }
}

// ---------------------------------------------------------------------------

extern "C" void kernel_launch(void* const* d_in, const int* in_sizes, int n_in,
                              void* d_out, int out_size, void* d_ws, size_t ws_size,
                              hipStream_t stream) {
  const float* hs    = (const float*)d_in[0];  // (2,2048,1024)
  const float* W_in  = (const float*)d_in[1];  // (1024,4096)
  const float* ck    = (const float*)d_in[2];  // (4,1,2048)
  const float* cb    = (const float*)d_in[3];  // (2048,)
  const float* W_x   = (const float*)d_in[4];  // (2048,96)
  const float* W_dt  = (const float*)d_in[5];  // (64,2048)
  const float* b_dt  = (const float*)d_in[6];  // (2048,)
  const float* A_log = (const float*)d_in[7];  // (2048,16)
  const float* Dp    = (const float*)d_in[8];  // (2048,)
  const float* W_out = (const float*)d_in[9];  // (2048,1024)
  float* out = (float*)d_out;                  // (2,2048,1024)

  char* ws = (char*)d_ws;
  auto alloc = [&](size_t bytes) -> char* {
    char* p = ws;
    ws += (bytes + 255) & ~(size_t)255;
    return p;
  };

  f16*   hs16   = (f16*)  alloc((size_t)MROWS * D_MODEL * 2);
  f16*   WxT    = (f16*)  alloc((size_t)D_INNER * D_MODEL * 2);
  f16*   WzT    = (f16*)  alloc((size_t)D_INNER * D_MODEL * 2);
  f16*   WxlT   = (f16*)  alloc((size_t)NXD * D_INNER * 2);
  f16*   WdtT   = (f16*)  alloc((size_t)D_INNER * DT_RANK * 2);
  f16*   WoutT  = (f16*)  alloc((size_t)D_MODEL * D_INNER * 2);
  float* xpre   = (float*)alloc((size_t)MROWS * D_INNER * 4);
  float* zf     = (float*)alloc((size_t)MROWS * D_INNER * 4);
  float* xf     = (float*)alloc((size_t)MROWS * D_INNER * 4);
  f16*   x16    = (f16*)  alloc((size_t)MROWS * D_INNER * 2);
  f16*   dt16   = (f16*)  alloc((size_t)MROWS * DT_RANK * 2);
  float* Bmat   = (float*)alloc((size_t)MROWS * D_STATE * 4);
  float* Cmat   = (float*)alloc((size_t)MROWS * D_STATE * 4);
  float* delta  = (float*)alloc((size_t)MROWS * D_INNER * 4);
  f16*   ypre16 = (f16*)  alloc((size_t)MROWS * D_INNER * 2);

  // Pack / transpose weights + activations to f16.
  k_pack_hs  <<<(MROWS * D_MODEL) / 256, 256, 0, stream>>>(hs, hs16, MROWS * D_MODEL);
  k_pack_win <<<(D_INNER * D_MODEL) / 256, 256, 0, stream>>>(W_in, WxT, WzT);
  k_pack_wx  <<<(NXD * D_INNER) / 256, 256, 0, stream>>>(W_x, WxlT);
  k_pack_wdt <<<(D_INNER * DT_RANK) / 256, 256, 0, stream>>>(W_dt, WdtT);
  k_pack_wout<<<(D_MODEL * D_INNER) / 256, 256, 0, stream>>>(W_out, WoutT);

  // GEMM 1a/1b: x_pre = hs @ Wx, z = hs @ Wz (deinterleaved columns).
  k_gemm_nt<<<dim3(MROWS / 64, D_INNER / 128), 256, 0, stream>>>(
      hs16, WxT, xpre, D_MODEL, D_INNER);
  k_gemm_nt<<<dim3(MROWS / 64, D_INNER / 128), 256, 0, stream>>>(
      hs16, WzT, zf, D_MODEL, D_INNER);

  // conv + SiLU.
  k_conv<<<(MROWS * D_INNER) / 256, 256, 0, stream>>>(xpre, ck, cb, xf, x16);

  // GEMM 2: x_dbl split.
  k_gemm_xdbl<<<MROWS / 128, 256, 0, stream>>>(x16, WxlT, dt16, Bmat, Cmat);

  // GEMM 3: delta = softplus(dt @ W_dt + b_dt).
  k_gemm_delta<<<dim3(MROWS / 128, D_INNER / 64), 256, 0, stream>>>(dt16, WdtT,
                                                                    b_dt, delta);
  // Selective scan (fused epilogue).
  k_scan<<<BATCH * (D_INNER / 256), 256, 0, stream>>>(delta, xf, Bmat, Cmat, zf,
                                                      A_log, Dp, ypre16);
  // GEMM 4: final projection.
  k_gemm_nt<<<dim3(MROWS / 64, D_MODEL / 128), 256, 0, stream>>>(
      ypre16, WoutT, out, D_INNER, D_MODEL);
}
